// Model_6717328851684
// MI455X (gfx1250) — matrix operand order repaired and checked by reading, not compiled, not run on hardware
//
#include <hip/hip_runtime.h>
#include <math.h>
#include <stdint.h>

// ---------------------------------------------------------------------------
// Autoformer forward on gfx1250 (MI455X).
// GEMMs run on v_wmma_f32_16x16x32_bf16 (bf16 in, fp32 accumulate).
// 64x128 block tile, 8 waves, LDS-staged with b128 fragment loads.
// Autocorrelation via packed-real radix-2 FFTs in LDS.
// ---------------------------------------------------------------------------

constexpr int BN   = 16;      // batch
constexpr int LN   = 1024;    // seq len (enc & dec)
constexpr int DN   = 512;     // d_model
constexpr int DFN  = 2048;    // d_ff
constexpr int CIN  = 21;      // enc_in / c_out
constexpr int CMK  = 4;       // mark dim
constexpr int NBL  = BN * LN; // 16384 rows
constexpr int TOPK = 20;      // int(3*log(1024))
constexpr int MAPAD = 12;     // (25-1)/2

typedef __attribute__((ext_vector_type(16))) __bf16 v16bf;
typedef __attribute__((ext_vector_type(8)))  float  v8f;
typedef __attribute__((ext_vector_type(8)))  unsigned short ushort8;
typedef unsigned short u16;

// ------------------------- small device helpers ----------------------------

__device__ __forceinline__ u16 f2bf(float f) {
  unsigned u = __float_as_uint(f);
  return (u16)((u + 0x7FFFu + ((u >> 16) & 1u)) >> 16);   // RNE
}

__device__ __forceinline__ float gelu_exact(float x) {
  return 0.5f * x * (1.0f + erff(x * 0.70710678118654752f));
}

// ------------------------------ convert ------------------------------------
// n must be a multiple of 8 (true for every tensor we convert).

__global__ void k_cvt_bf16(const float* __restrict__ src, u16* __restrict__ dst, size_t n8) {
  size_t i = (size_t)blockIdx.x * blockDim.x + threadIdx.x;
  if (i >= n8) return;
  const float* s = src + i * 8;
  ushort8 o;
  #pragma unroll
  for (int j = 0; j < 8; ++j) o[j] = f2bf(s[j]);
  *(ushort8*)(dst + i * 8) = o;
}

// ------------------------------- WMMA GEMM ---------------------------------
// C[M,N] = A[M,K] x B[K,N]   (bf16 inputs, fp32 accumulate)
// flags: 1 = +bias[N], 2 = exact GELU, 4 = +residual[M,N]
// M % 64 == 0, N % 128 == 0, K % 32 == 0.
// Block: 256 threads = 8 waves. Tile 64(M) x 128(N) x 32(K).
// Wave (wv&3) -> 16-row M slice, (wv>>2) -> 64-col N slice = 4 WMMA frags.
// LDS: A row-major [64][40] (stride 40 u16 = 80B, 16B aligned, bank-spread),
//      B TRANSPOSED  [128][40] so per-lane fragments are contiguous b128 pairs.

#define GF_BIAS 1
#define GF_GELU 2
#define GF_RES  4

__global__ __launch_bounds__(256) void k_gemm_wmma(
    const u16* __restrict__ A, const u16* __restrict__ Bm,
    const float* __restrict__ bias, const float* __restrict__ res,
    float* __restrict__ C, int M, int N, int K, int flags)
{
  __shared__ u16 As[64 * 40];
  __shared__ u16 Bs[128 * 40];

  const int m0   = blockIdx.y * 64;
  const int n0   = blockIdx.x * 128;
  const int tid  = threadIdx.x;
  const int lane = tid & 31;
  const int wv   = tid >> 5;          // 8 waves
  const int mw   = (wv & 3) * 16;     // wave M sub-tile
  const int nw   = (wv >> 2) * 64;    // wave N sub-tile (4 x 16)
  const int half = lane >> 4;
  const int lm   = lane & 15;

  // staging coordinates
  const int ar  = tid >> 2;           // 0..63   A row
  const int ac  = (tid & 3) * 8;      // 0,8,16,24
  const int bkk = tid >> 4;           // 0..15   B k (chunk 0); +16 for chunk 1
  const int bn8 = (tid & 15) * 8;     // 0..120  B n base

  v8f acc[4] = {v8f{}, v8f{}, v8f{}, v8f{}};

  for (int k0 = 0; k0 < K; k0 += 32) {
    // ---- global loads (b128 each) ----
    const ushort8 av  = *(const ushort8*)&A [(size_t)(m0 + ar) * K + (k0 + ac)];
    const ushort8 bv0 = *(const ushort8*)&Bm[(size_t)(k0 + bkk)      * N + (n0 + bn8)];
    const ushort8 bv1 = *(const ushort8*)&Bm[(size_t)(k0 + bkk + 16) * N + (n0 + bn8)];
    if (k0 + 32 < K) {   // prefetch next K tile (global_prefetch_b8)
      __builtin_prefetch(&A [(size_t)(m0 + ar) * K + (k0 + 32 + ac)], 0, 3);
      __builtin_prefetch(&Bm[(size_t)(k0 + 32 + bkk) * N + (n0 + bn8)], 0, 3);
    }

    // ---- LDS stores: A row-major b128, B transposed b16 scatter ----
    *(ushort8*)&As[ar * 40 + ac] = av;
    #pragma unroll
    for (int j = 0; j < 8; ++j) Bs[(bn8 + j) * 40 + bkk]      = bv0[j];
    #pragma unroll
    for (int j = 0; j < 8; ++j) Bs[(bn8 + j) * 40 + bkk + 16] = bv1[j];
    __syncthreads();

    // ---- fragments: contiguous b128 pairs ----
    union Frag { ushort8 h[2]; v16bf v; } fa, fb;
    const int arow = (mw + lm) * 40;
    fa.h[0] = *(const ushort8*)&As[arow + half * 8];        // K 0..7 / 8..15
    fa.h[1] = *(const ushort8*)&As[arow + 16 + half * 8];   // K 16..23 / 24..31
    #pragma unroll
    for (int f = 0; f < 4; ++f) {
      const int brow = (nw + f * 16 + lm) * 40;
      fb.h[0] = *(const ushort8*)&Bs[brow + half * 8];
      fb.h[1] = *(const ushort8*)&Bs[brow + 16 + half * 8];
      acc[f] = __builtin_amdgcn_wmma_f32_16x16x32_bf16(false, fa.v, false, fb.v,
                                                       (short)0, acc[f], false, false);
    }
    __syncthreads();
  }

  // ---- epilogue ----
  #pragma unroll
  for (int f = 0; f < 4; ++f) {
    #pragma unroll
    for (int r = 0; r < 8; ++r) {
      const int m = m0 + mw + half * 8 + r;
      const int n = n0 + nw + f * 16 + lm;
      float v = acc[f][r];
      if (flags & GF_BIAS) v += bias[n];
      if (flags & GF_GELU) v = gelu_exact(v);
      const size_t idx = (size_t)m * N + n;
      if (flags & GF_RES) v += res[idx];
      C[idx] = v;
    }
  }
}

// -------------------------- embedding (circ conv) --------------------------
// out[b,t,d] = sum_c sum_j x[b,(t-1+j)%L,c]*conv[d,c,j] + sum_m mark[b,t,m]*markW[m,d]

__global__ void k_embed(const float* __restrict__ x, const float* __restrict__ mark,
                        const float* __restrict__ convW, const float* __restrict__ markW,
                        float* __restrict__ out)
{
  size_t i = (size_t)blockIdx.x * blockDim.x + threadIdx.x;
  if (i >= (size_t)NBL * DN) return;
  const int d = (int)(i & (DN - 1));
  const int t = (int)((i >> 9) & (LN - 1));
  const int b = (int)(i >> 19);
  float s = 0.f;
  #pragma unroll
  for (int j = 0; j < 3; ++j) {
    const int tj = (t - 1 + j + LN) & (LN - 1);
    const float* xr = &x[((size_t)(b * LN + tj)) * CIN];
    const float* wr = &convW[(size_t)d * CIN * 3 + j];
    for (int c = 0; c < CIN; ++c) s += xr[c] * wr[c * 3];
  }
  const float* mr = &mark[((size_t)(b * LN + t)) * CMK];
  #pragma unroll
  for (int m = 0; m < CMK; ++m) s += mr[m] * markW[m * DN + d];
  out[i] = s;
}

// --------------------------- series decomposition --------------------------
// moving average window 25 with replicate padding == index clamping.
// trend_mode: 0 = discard trend, 1 = write trend, 2 = trend += (accumulate)

__global__ void k_decomp(const float* __restrict__ x, float* __restrict__ season,
                         float* __restrict__ trend, int Bn, int Ln, int Dn, int trend_mode)
{
  size_t i = (size_t)blockIdx.x * blockDim.x + threadIdx.x;
  size_t total = (size_t)Bn * Ln * Dn;
  if (i >= total) return;
  const int d = (int)(i % Dn);
  const int t = (int)((i / Dn) % Ln);
  const int b = (int)(i / ((size_t)Dn * Ln));
  float s = 0.f;
  #pragma unroll
  for (int j = -MAPAD; j <= MAPAD; ++j) {
    int tj = t + j;
    tj = tj < 0 ? 0 : (tj >= Ln ? Ln - 1 : tj);
    s += x[((size_t)b * Ln + tj) * Dn + d];
  }
  const float m = s * (1.0f / 25.0f);
  season[i] = x[i] - m;
  if (trend_mode == 1)      trend[i]  = m;
  else if (trend_mode == 2) trend[i] += m;
}

// ------------------------------ FFT pipeline -------------------------------

__device__ __forceinline__ void fft1024(float2* z, int tid) {
  for (int s = 1; s <= 10; ++s) {
    const int m = 1 << s, hm = m >> 1;
    for (int i = tid; i < 512; i += 256) {
      const int pos = i & (hm - 1);
      const int idx = ((i >> (s - 1)) << s) + pos;
      float sw, cw;
      __sincosf(-6.2831853071795865f * (float)pos / (float)m, &sw, &cw);
      const float2 u = z[idx];
      const float2 v = z[idx + hm];
      const float tr = cw * v.x - sw * v.y;
      const float ti = cw * v.y + sw * v.x;
      z[idx]      = make_float2(u.x + tr, u.y + ti);
      z[idx + hm] = make_float2(u.x - tr, u.y - ti);
    }
    __syncthreads();
  }
}

// one block per (b, channel): packed real FFT of q & k, accumulate Q*conj(K)
__global__ __launch_bounds__(256) void k_fft_corr(const float* __restrict__ Q,
                                                  const float* __restrict__ Kv,
                                                  float* __restrict__ acc)
{
  __shared__ float2 z[1024];
  const int b  = blockIdx.x >> 9;
  const int ch = blockIdx.x & 511;
  const int tid = threadIdx.x;
  for (int t = tid; t < 1024; t += 256) {
    const int rev = (int)(__brev((unsigned)t) >> 22);
    z[rev] = make_float2(Q[((size_t)(b * LN + t)) * DN + ch],
                         Kv[((size_t)(b * LN + t)) * DN + ch]);
  }
  __syncthreads();
  fft1024(z, tid);
  for (int f = tid; f < 1024; f += 256) {
    const float2 Zf = z[f];
    const float2 Zn = z[(1024 - f) & 1023];
    const float cr = Zn.x, ci = -Zn.y;           // conj(Z[N-f])
    const float qr = 0.5f * (Zf.x + cr), qi = 0.5f * (Zf.y + ci);
    const float dr = Zf.x - cr,          di = Zf.y - ci;
    const float kr = 0.5f * di,          ki = -0.5f * dr;  // (Zf-conjZn)/(2i)
    const float pr = qr * kr + qi * ki;          // Q * conj(K)
    const float pi = qi * kr - qr * ki;
    atomicAdd(&acc[((size_t)(b * 1024 + f)) * 2 + 0], pr);
    atomicAdd(&acc[((size_t)(b * 1024 + f)) * 2 + 1], pi);
  }
}

// one block per batch: ifft(acc)/L, /(H*E) for channel mean
__global__ __launch_bounds__(256) void k_ifft_mean(const float* __restrict__ acc,
                                                   float* __restrict__ mc)
{
  __shared__ float2 z[1024];
  const int b = blockIdx.x, tid = threadIdx.x;
  for (int f = tid; f < 1024; f += 256) {
    const int rev = (int)(__brev((unsigned)f) >> 22);
    z[rev] = make_float2(acc[((size_t)(b * 1024 + f)) * 2 + 0],
                         -acc[((size_t)(b * 1024 + f)) * 2 + 1]);  // conj
  }
  __syncthreads();
  fft1024(z, tid);
  const float scale = 1.0f / (1024.0f * 512.0f);
  for (int t = tid; t < 1024; t += 256)
    mc[b * 1024 + t] = z[t].x * scale;           // Re(conj(fft)) = Re(fft)
}

// per-batch top-20 + softmax
__global__ __launch_bounds__(256) void k_topk(const float* __restrict__ mc,
                                              int* __restrict__ dly,
                                              float* __restrict__ wts)
{
  __shared__ float buf[1024];
  __shared__ float rv[256];
  __shared__ int   ri[256];
  __shared__ float selv[TOPK];
  __shared__ int   seli[TOPK];
  const int b = blockIdx.x, tid = threadIdx.x;
  for (int t = tid; t < 1024; t += 256) buf[t] = mc[b * 1024 + t];
  __syncthreads();
  for (int kk = 0; kk < TOPK; ++kk) {
    float best = -1e30f; int bi = 0;
    for (int t = tid; t < 1024; t += 256) {
      const float v = buf[t];
      if (v > best) { best = v; bi = t; }
    }
    rv[tid] = best; ri[tid] = bi;
    __syncthreads();
    for (int off = 128; off > 0; off >>= 1) {
      if (tid < off && rv[tid + off] > rv[tid]) { rv[tid] = rv[tid + off]; ri[tid] = ri[tid + off]; }
      __syncthreads();
    }
    if (tid == 0) { selv[kk] = rv[0]; seli[kk] = ri[0]; buf[ri[0]] = -1e30f; }
    __syncthreads();
  }
  if (tid == 0) {
    const float mx = selv[0];
    float e[TOPK], sum = 0.f;
    for (int i = 0; i < TOPK; ++i) { e[i] = __expf(selv[i] - mx); sum += e[i]; }
    const float inv = 1.0f / sum;
    for (int i = 0; i < TOPK; ++i) { wts[b * 32 + i] = e[i] * inv; dly[b * 32 + i] = seli[i]; }
  }
}

// agg[b,t,d] = sum_i w[b,i] * V[b,(t+delay_i)%L,d]
__global__ void k_agg(const float* __restrict__ V, const int* __restrict__ dly,
                      const float* __restrict__ wts, float* __restrict__ out)
{
  size_t i = (size_t)blockIdx.x * blockDim.x + threadIdx.x;
  if (i >= (size_t)NBL * DN) return;
  const int d = (int)(i & (DN - 1));
  const int t = (int)((i >> 9) & (LN - 1));
  const int b = (int)(i >> 19);
  float s = 0.f;
  #pragma unroll
  for (int j = 0; j < TOPK; ++j) {
    const int   dl = dly[b * 32 + j];
    const float w  = wts[b * 32 + j];
    const int   tt = (t + dl) & (LN - 1);
    s += w * V[((size_t)(b * LN + tt)) * DN + d];
  }
  out[i] = s;
}

// ----------------------------- my_layernorm --------------------------------

__global__ __launch_bounds__(256) void k_ln_row(const float* __restrict__ x,
                                                const float* __restrict__ g,
                                                const float* __restrict__ bb,
                                                float* __restrict__ out)
{
  __shared__ float red[256];
  const int row = blockIdx.x, tid = threadIdx.x;
  const float* xr = &x[(size_t)row * DN];
  float s = 0.f;
  for (int d = tid; d < DN; d += 256) s += xr[d];
  red[tid] = s; __syncthreads();
  for (int off = 128; off > 0; off >>= 1) { if (tid < off) red[tid] += red[tid + off]; __syncthreads(); }
  const float mu = red[0] / DN;
  __syncthreads();
  float v = 0.f;
  for (int d = tid; d < DN; d += 256) { const float e = xr[d] - mu; v += e * e; }
  red[tid] = v; __syncthreads();
  for (int off = 128; off > 0; off >>= 1) { if (tid < off) red[tid] += red[tid + off]; __syncthreads(); }
  const float rstd = rsqrtf(red[0] / DN + 1e-5f);
  for (int d = tid; d < DN; d += 256)
    out[(size_t)row * DN + d] = (xr[d] - mu) * rstd * g[d] + bb[d];
}

// subtract per-(b,d) mean over time
__global__ void k_ln_colsub(const float* __restrict__ xin, float* __restrict__ xout)
{
  const int i = blockIdx.x * blockDim.x + threadIdx.x;
  if (i >= BN * DN) return;
  const int d = i & (DN - 1), b = i >> 9;
  float s = 0.f;
  for (int t = 0; t < LN; ++t) s += xin[((size_t)(b * LN + t)) * DN + d];
  const float mu = s / LN;
  for (int t = 0; t < LN; ++t) {
    const size_t idx = ((size_t)(b * LN + t)) * DN + d;
    xout[idx] = xin[idx] - mu;
  }
}

// --------------------------- decoder init / trend --------------------------

__global__ void k_meantime(const float* __restrict__ x, float* __restrict__ out)
{
  const int i = blockIdx.x * blockDim.x + threadIdx.x;
  if (i >= BN * CIN) return;
  const int c = i % CIN, b = i / CIN;
  float s = 0.f;
  for (int t = 0; t < LN; ++t) s += x[((size_t)(b * LN + t)) * CIN + c];
  out[i] = s / LN;
}

// seasonal_init / trend_init (== T initial): tails identity, middle zeros/mean
__global__ void k_buildinit(const float* __restrict__ seasx, const float* __restrict__ trendx,
                            const float* __restrict__ meanx, float* __restrict__ sinit,
                            float* __restrict__ T)
{
  const int i = blockIdx.x * blockDim.x + threadIdx.x;
  if (i >= BN * LN * CIN) return;
  const int c = i % CIN;
  const int t = (i / CIN) % LN;
  const int b = i / (CIN * LN);
  const bool mid = (t >= 256) && (t < 768);
  sinit[i] = mid ? 0.f : seasx[i];
  T[i]     = mid ? meanx[b * CIN + c] : trendx[i];
}

// T[b,t,c] += sum_d sum_j acc[b,(t-1+j)%L,d] * wtrend[c,d,j]
__global__ void k_trendconv(const float* __restrict__ acc, const float* __restrict__ wt,
                            float* __restrict__ T)
{
  const int i = blockIdx.x * blockDim.x + threadIdx.x;
  if (i >= BN * LN * CIN) return;
  const int c = i % CIN;
  const int t = (i / CIN) % LN;
  const int b = i / (CIN * LN);
  float s = 0.f;
  #pragma unroll
  for (int j = 0; j < 3; ++j) {
    const int tj = (t - 1 + j + LN) & (LN - 1);
    const float* ar = &acc[((size_t)(b * LN + tj)) * DN];
    const float* wr = &wt[(size_t)c * DN * 3 + j];
    for (int d = 0; d < DN; ++d) s += ar[d] * wr[d * 3];
  }
  T[i] += s;
}

// out[b,t',c] = T[b,t'+256,c] + dec[b,t'+256,:] @ projw[:,c] + projb[c]
__global__ void k_projout(const float* __restrict__ dec, const float* __restrict__ T,
                          const float* __restrict__ pw, const float* __restrict__ pb,
                          float* __restrict__ out)
{
  const int i = blockIdx.x * blockDim.x + threadIdx.x;
  if (i >= BN * 512 * CIN) return;
  const int c  = i % CIN;
  const int tp = (i / CIN) % 512;
  const int b  = i / (CIN * 512);
  const int t  = tp + 256;
  const float* dr = &dec[((size_t)(b * LN + t)) * DN];
  float s = pb[c];
  for (int d = 0; d < DN; ++d) s += dr[d] * pw[d * CIN + c];
  out[i] = T[((size_t)(b * LN + t)) * CIN + c] + s;
}

// ------------------------------- host side ---------------------------------

namespace {

struct Ctx {
  hipStream_t st;
  u16* Abf;   // NBL x 2048 bf16
  u16* Wbf;   // 2048 x 512 bf16
  float* corr;   // B x 1024 x 2
  float* mcorr;  // B x 1024
  int*   dly;    // B x 32
  float* wts;    // B x 32
};

inline void launch_cvt(hipStream_t st, const float* src, u16* dst, size_t n) {
  const size_t n8 = n / 8;
  k_cvt_bf16<<<dim3((unsigned)((n8 + 255) / 256)), dim3(256), 0, st>>>(src, dst, n8);
}

inline void launch_gemm(hipStream_t st, const u16* A, const u16* W, const float* bias,
                        const float* res, float* out, int M, int N, int K, int flags) {
  k_gemm_wmma<<<dim3((unsigned)(N / 128), (unsigned)(M / 64)), dim3(256), 0, st>>>(
      A, W, bias, res, out, M, N, K, flags);
}

struct AttnW { const float *wq, *bq, *wk, *bk, *wv, *bv, *wo, *bo; };

// out = resid + attn(xq, xkv); uses S1,S2,S3 (q,k,v) and S4 (agg)
inline void run_attn(const Ctx& c, const float* xq, const float* xkv, const AttnW& aw,
                     const float* resid, float* out,
                     float* S1, float* S2, float* S3, float* S4) {
  launch_cvt(c.st, xq, c.Abf, (size_t)NBL * DN);
  launch_cvt(c.st, aw.wq, c.Wbf, (size_t)DN * DN);
  launch_gemm(c.st, c.Abf, c.Wbf, aw.bq, nullptr, S1, NBL, DN, DN, GF_BIAS);
  if (xkv != xq) launch_cvt(c.st, xkv, c.Abf, (size_t)NBL * DN);
  launch_cvt(c.st, aw.wk, c.Wbf, (size_t)DN * DN);
  launch_gemm(c.st, c.Abf, c.Wbf, aw.bk, nullptr, S2, NBL, DN, DN, GF_BIAS);
  launch_cvt(c.st, aw.wv, c.Wbf, (size_t)DN * DN);
  launch_gemm(c.st, c.Abf, c.Wbf, aw.bv, nullptr, S3, NBL, DN, DN, GF_BIAS);

  hipMemsetAsync(c.corr, 0, (size_t)BN * 1024 * 2 * sizeof(float), c.st);
  k_fft_corr<<<dim3(BN * 512), dim3(256), 0, c.st>>>(S1, S2, c.corr);
  k_ifft_mean<<<dim3(BN), dim3(256), 0, c.st>>>(c.corr, c.mcorr);
  k_topk<<<dim3(BN), dim3(256), 0, c.st>>>(c.mcorr, c.dly, c.wts);
  k_agg<<<dim3((unsigned)(((size_t)NBL * DN + 255) / 256)), dim3(256), 0, c.st>>>(
      S3, c.dly, c.wts, S4);

  launch_cvt(c.st, S4, c.Abf, (size_t)NBL * DN);
  launch_cvt(c.st, aw.wo, c.Wbf, (size_t)DN * DN);
  launch_gemm(c.st, c.Abf, c.Wbf, aw.bo, resid, out, NBL, DN, DN, GF_BIAS | GF_RES);
}

// out = x + gelu(x@w1)@w2 ; uses H as hidden scratch
inline void run_ffn(const Ctx& c, const float* x, const float* w1, const float* w2,
                    float* H, float* out) {
  launch_cvt(c.st, x, c.Abf, (size_t)NBL * DN);
  launch_cvt(c.st, w1, c.Wbf, (size_t)DN * DFN);
  launch_gemm(c.st, c.Abf, c.Wbf, nullptr, nullptr, H, NBL, DFN, DN, GF_GELU);
  launch_cvt(c.st, H, c.Abf, (size_t)NBL * DFN);
  launch_cvt(c.st, w2, c.Wbf, (size_t)DFN * DN);
  launch_gemm(c.st, c.Abf, c.Wbf, nullptr, x, out, NBL, DN, DFN, GF_RES);
}

} // namespace

extern "C" void kernel_launch(void* const* d_in, const int* in_sizes, int n_in,
                              void* d_out, int out_size, void* d_ws, size_t ws_size,
                              hipStream_t stream) {
  (void)in_sizes; (void)n_in; (void)out_size; (void)ws_size;

  // -------- inputs (jax pytree sorted-key flattening order) --------
  const float* x_enc      = (const float*)d_in[0];
  const float* x_mark_enc = (const float*)d_in[1];
  // d_in[2] = x_dec (values unused by the model)
  const float* x_mark_dec = (const float*)d_in[3];
  const float* dec_conv = (const float*)d_in[4];
  const float* dec_mark = (const float*)d_in[5];
  AttnW cross { (const float*)d_in[12], (const float*)d_in[8],   // wq,bq
                (const float*)d_in[10], (const float*)d_in[6],   // wk,bk
                (const float*)d_in[13], (const float*)d_in[9],   // wv,bv
                (const float*)d_in[11], (const float*)d_in[7] }; // wo,bo
  AttnW dself { (const float*)d_in[20], (const float*)d_in[16],
                (const float*)d_in[18], (const float*)d_in[14],
                (const float*)d_in[21], (const float*)d_in[17],
                (const float*)d_in[19], (const float*)d_in[15] };
  const float* dec_w1     = (const float*)d_in[22];
  const float* dec_w2     = (const float*)d_in[23];
  const float* dec_wtrend = (const float*)d_in[24];
  const float* dec_ln_b   = (const float*)d_in[25];
  const float* dec_ln_g   = (const float*)d_in[26];
  const float* enc_conv   = (const float*)d_in[27];
  const float* enc_mark   = (const float*)d_in[28];
  AttnW encA[2];
  for (int l = 0; l < 2; ++l) {
    const int base = 29 + l * 10;
    encA[l] = AttnW{ (const float*)d_in[base + 6], (const float*)d_in[base + 2],
                     (const float*)d_in[base + 4], (const float*)d_in[base + 0],
                     (const float*)d_in[base + 7], (const float*)d_in[base + 3],
                     (const float*)d_in[base + 5], (const float*)d_in[base + 1] };
  }
  const float* enc_w1[2] = { (const float*)d_in[37], (const float*)d_in[47] };
  const float* enc_w2[2] = { (const float*)d_in[38], (const float*)d_in[48] };
  const float* enc_ln_b  = (const float*)d_in[49];
  const float* enc_ln_g  = (const float*)d_in[50];
  const float* proj_b    = (const float*)d_in[51];
  const float* proj_w    = (const float*)d_in[52];

  // -------- workspace carving --------
  uint8_t* p = (uint8_t*)d_ws;
  auto carve = [&](size_t bytes) -> void* {
    void* r = (void*)p;
    p += (bytes + 255) & ~(size_t)255;
    return r;
  };
  const size_t BD = (size_t)NBL * DN * sizeof(float);
  float* S0 = (float*)carve(BD);               // enc stream / enc_out
  float* S1 = (float*)carve(BD);               // q
  float* S2 = (float*)carve(BD);               // k
  float* S3 = (float*)carve(BD);               // v
  float* S4 = (float*)carve(BD);               // agg / LN temp
  float* S5 = (float*)carve(BD);               // x + sublayer
  float* S6 = (float*)carve(BD);               // dec stream
  float* S7 = (float*)carve(BD);               // dec trend accumulator (D_MODEL)
  float* Hh = (float*)carve((size_t)NBL * DFN * sizeof(float));
  u16* Abf  = (u16*)carve((size_t)NBL * DFN * sizeof(u16));
  u16* Wbf  = (u16*)carve((size_t)DFN * DN * sizeof(u16));
  float* corr  = (float*)carve((size_t)BN * 1024 * 2 * sizeof(float));
  float* mcorr = (float*)carve((size_t)BN * 1024 * sizeof(float));
  int*   dly   = (int*)carve((size_t)BN * 32 * sizeof(int));
  float* wts   = (float*)carve((size_t)BN * 32 * sizeof(float));
  float* seasx = (float*)carve((size_t)BN * LN * CIN * sizeof(float));
  float* trendx= (float*)carve((size_t)BN * LN * CIN * sizeof(float));
  float* meanx = (float*)carve((size_t)BN * CIN * sizeof(float));
  float* sinit = (float*)carve((size_t)BN * LN * CIN * sizeof(float));
  float* Ttr   = (float*)carve((size_t)BN * LN * CIN * sizeof(float));

  Ctx c{stream, Abf, Wbf, corr, mcorr, dly, wts};

  const unsigned nBLD  = (unsigned)(((size_t)NBL * DN + 255) / 256);
  const unsigned nBLC  = (unsigned)((BN * LN * CIN + 255) / 256);

  // ============================ encoder ============================
  k_embed<<<dim3(nBLD), dim3(256), 0, stream>>>(x_enc, x_mark_enc, enc_conv, enc_mark, S0);
  for (int l = 0; l < 2; ++l) {
    run_attn(c, S0, S0, encA[l], /*resid=*/S0, /*out=*/S5, S1, S2, S3, S4);
    k_decomp<<<dim3(nBLD), dim3(256), 0, stream>>>(S5, S0, nullptr, BN, LN, DN, 0);
    run_ffn(c, S0, enc_w1[l], enc_w2[l], Hh, S5);
    k_decomp<<<dim3(nBLD), dim3(256), 0, stream>>>(S5, S0, nullptr, BN, LN, DN, 0);
  }
  k_ln_row<<<dim3(NBL), dim3(256), 0, stream>>>(S0, enc_ln_g, enc_ln_b, S4);
  k_ln_colsub<<<dim3((BN * DN + 255) / 256), dim3(256), 0, stream>>>(S4, S0); // S0 = enc_out

  // ========================= decoder init ==========================
  k_decomp<<<dim3(nBLC), dim3(256), 0, stream>>>(x_enc, seasx, trendx, BN, LN, CIN, 1);
  k_meantime<<<dim3((BN * CIN + 255) / 256), dim3(256), 0, stream>>>(x_enc, meanx);
  k_buildinit<<<dim3(nBLC), dim3(256), 0, stream>>>(seasx, trendx, meanx, sinit, Ttr);
  k_embed<<<dim3(nBLD), dim3(256), 0, stream>>>(sinit, x_mark_dec, dec_conv, dec_mark, S6);
  hipMemsetAsync(S7, 0, BD, stream);

  // ========================= decoder layer =========================
  run_attn(c, S6, S6, dself, /*resid=*/S6, /*out=*/S5, S1, S2, S3, S4);
  k_decomp<<<dim3(nBLD), dim3(256), 0, stream>>>(S5, S6, S7, BN, LN, DN, 2);
  run_attn(c, S6, /*xkv=*/S0, cross, /*resid=*/S6, /*out=*/S5, S1, S2, S3, S4);
  k_decomp<<<dim3(nBLD), dim3(256), 0, stream>>>(S5, S6, S7, BN, LN, DN, 2);
  run_ffn(c, S6, dec_w1, dec_w2, Hh, S5);
  k_decomp<<<dim3(nBLD), dim3(256), 0, stream>>>(S5, S6, S7, BN, LN, DN, 2);
  k_trendconv<<<dim3(nBLC), dim3(256), 0, stream>>>(S7, dec_wtrend, Ttr);

  k_ln_row<<<dim3(NBL), dim3(256), 0, stream>>>(S6, dec_ln_g, dec_ln_b, S4);
  k_ln_colsub<<<dim3((BN * DN + 255) / 256), dim3(256), 0, stream>>>(S4, S6);

  // ===================== projection + slice out ====================
  k_projout<<<dim3((BN * 512 * CIN + 255) / 256), dim3(256), 0, stream>>>(
      S6, Ttr, proj_w, proj_b, (float*)d_out);
}
